// DecodingAttentionWrapper_3066606649823
// MI455X (gfx1250) — matrix operand order repaired and checked by reading, not compiled, not run on hardware
//
#include <hip/hip_runtime.h>
#include <hip/hip_fp16.h>
#include <math.h>

#define B_   8
#define H_   32
#define D_   128
#define T_   64
#define P_   64
#define NB_  512
#define SEL_ 32

typedef __attribute__((ext_vector_type(16))) _Float16     v16h;
typedef __attribute__((ext_vector_type(8)))  _Float16     v8h;
typedef __attribute__((ext_vector_type(8)))  float        v8f;
typedef __attribute__((ext_vector_type(4)))  unsigned int u32x4;
typedef __attribute__((ext_vector_type(8)))  int          i32x8;
typedef __attribute__((ext_vector_type(4)))  int          i32x4;

#if __has_builtin(__builtin_amdgcn_tensor_load_to_lds) && \
    __has_builtin(__builtin_amdgcn_s_wait_tensorcnt)
#define HAVE_TDM 1
#else
#define HAVE_TDM 0
#endif

// ---------------------------------------------------------------------------
// Kernel 1: per-(block,head) mean of k_cache over T.  Pure streaming, 512 MB.
// ---------------------------------------------------------------------------
__global__ __launch_bounds__(128) void page_mean_kernel(
    const int* __restrict__ kc, float* __restrict__ pm)
{
    int bh = blockIdx.x;
    int d  = threadIdx.x;
    const int* base = kc + (size_t)bh * (T_ * D_);
    int s = 0;
#pragma unroll 8
    for (int t = 0; t < T_; ++t) s += base[t * D_ + d];
    pm[(size_t)bh * D_ + d] = (float)s * (1.0f / (float)T_);
}

// ---------------------------------------------------------------------------
// Kernel 2: RoPE(q) + page scores + rank-based top-32 selection.
// ---------------------------------------------------------------------------
__global__ __launch_bounds__(128) void select_kernel(
    const float* __restrict__ q, const float* __restrict__ pm,
    const int* __restrict__ btab, const int* __restrict__ seqlens,
    float* __restrict__ qr_out, int* __restrict__ sel_out)
{
    __shared__ __align__(16) float qr_s[D_];
    __shared__ float sc[P_];

    int b = blockIdx.x / H_;
    int h = blockIdx.x % H_;
    int tid = threadIdx.x;
    int seqlen = seqlens[b];

    {
        int i = tid & 63;
        float inv = __expf(-(float)i * (9.210340371976184f / 64.0f));
        float ang = (float)(seqlen - 1) * inv;
        float s, c;
        __sincosf(ang, &s, &c);
        const float* qv = q + ((size_t)b * H_ + h) * D_;
        float v;
        if (tid < 64) v = qv[tid] * c - qv[tid + 64] * s;
        else          v = qv[tid] * c + qv[tid - 64] * s;
        qr_s[tid] = v;
        qr_out[((size_t)b * H_ + h) * D_ + tid] = v;
    }
    __syncthreads();

    int nvalid = (seqlen + T_ - 1) / T_;
    if (tid < P_) {
        float acc;
        if (tid < nvalid) {
            int phys = btab[b * P_ + tid];
            const float4* row = (const float4*)(pm + ((size_t)phys * H_ + h) * D_);
            const float4* q4  = (const float4*)qr_s;
            acc = 0.0f;
#pragma unroll 8
            for (int j = 0; j < D_ / 4; ++j) {
                float4 a = row[j]; float4 qq = q4[j];
                acc += a.x * qq.x + a.y * qq.y + a.z * qq.z + a.w * qq.w;
            }
        } else {
            acc = -__builtin_inff();
        }
        sc[tid] = acc;
    }
    __syncthreads();
    if (tid < P_) {
        float mys = sc[tid];
        int rank = 0;
#pragma unroll 8
        for (int p = 0; p < P_; ++p) {
            float o = sc[p];
            rank += ((o > mys) || (o == mys && p < tid)) ? 1 : 0;
        }
        if (rank < SEL_ && mys != -__builtin_inff())
            sel_out[((size_t)b * H_ + h) * SEL_ + rank] = tid;
    }
}

// ---------------------------------------------------------------------------
// TDM: DMA one 32 KB page (8192 x int32, contiguous) from global into LDS.
// D# per CDNA5 ISA: 1-row 2D tile, data_size = 4 B.
// ---------------------------------------------------------------------------
#if HAVE_TDM
__device__ __forceinline__ void tdm_load_page(const int* gsrc, void* ldsdst)
{
    unsigned long long ga = (unsigned long long)(uintptr_t)gsrc;
    unsigned int       la = (unsigned int)(uintptr_t)ldsdst;
    const unsigned int N  = (unsigned int)(T_ * D_);   // 8192 elements

    u32x4 g0;
    g0[0] = 1u;                                        // count=1, no gather
    g0[1] = la;                                        // lds_addr
    g0[2] = (unsigned int)ga;                          // global_addr[31:0]
    g0[3] = (unsigned int)((ga >> 32) & 0x01FFFFFFu)   // global_addr[56:32]
          | 0x80000000u;                               // type=2 ("image")

    i32x8 g1;
    g1[0] = (int)(2u << 16);                           // wg_mask=0, data_size=4B
    g1[1] = (int)((N & 0xFFFFu) << 16);                // tensor_dim0[15:0]
    g1[2] = (int)((N >> 16) | (1u << 16));             // tensor_dim0[31:16], tensor_dim1=1
    g1[3] = (int)(N << 16);                            // tile_dim0 = 8192
    g1[4] = 1;                                         // tile_dim1 = 1, tile_dim2 = 0
    g1[5] = (int)N;                                    // tensor_dim0_stride[31:0]
    g1[6] = 0;                                         // stride hi, dim1_stride lo
    g1[7] = 0;

    i32x4 g2 = {0, 0, 0, 0};
    i32x4 g3 = {0, 0, 0, 0};
#if defined(__clang_major__) && (__clang_major__ >= 23)
    i32x8 g4 = {0, 0, 0, 0, 0, 0, 0, 0};
    __builtin_amdgcn_tensor_load_to_lds(g0, g1, g2, g3, g4, 0);
#else
    __builtin_amdgcn_tensor_load_to_lds(g0, g1, g2, g3, 0);
#endif
}
#endif

// Fallback / portable staging: all 128 threads copy the page into LDS.
__device__ __forceinline__ void vgpr_stage_page(const int* gsrc, int* ldsdst, int tid)
{
    const int4* src = (const int4*)gsrc;
#pragma unroll 4
    for (int it = 0; it < 16; ++it) {
        int idx = it * 128 + tid;
        *(int4*)&ldsdst[idx * 4] = src[idx];
    }
}

// ---------------------------------------------------------------------------
// Kernel 3: attention over 32 selected pages.
// QK on v_wmma_f32_16x16x32_f16; K/V pages streamed by the Tensor Data Mover
// into double-buffered LDS (raw int32), converted to f16 at fragment build.
// grid = B*H blocks, 128 threads (4 waves; wave w owns token tile w).
// ---------------------------------------------------------------------------
__global__ __launch_bounds__(128) void attn_kernel(
    const int* __restrict__ kc, const int* __restrict__ vc,
    const int* __restrict__ btab, const int* __restrict__ seqlens,
    const float* __restrict__ qr, const int* __restrict__ sel,
    const float* __restrict__ kscale_p, const float* __restrict__ vscale_p,
    float* __restrict__ out)
{
    __shared__ __align__(16) int kvbuf[2][T_ * D_];    // 2 x 32 KB double buffer
    __shared__ float att[SEL_ * T_];                   // 8 KB scores
    __shared__ _Float16 qh[D_];
    __shared__ float red[128];

    int b = blockIdx.x / H_;
    int h = blockIdx.x % H_;
    int tid  = threadIdx.x;
    int lane = tid & 31;
    int wave = tid >> 5;
    int seqlen = seqlens[b];
    float vscale = vscale_p[0];
    float score_scale = kscale_p[0] * rsqrtf((float)D_);

    qh[tid] = (_Float16)qr[((size_t)b * H_ + h) * D_ + tid];
    __syncthreads();

    // B matrices: column 0 = q chunk.  B 32x16 f16 layout: lanes 0-15 hold
    // K{0-7,16-23} of their column, lanes 16-31 hold K{8-15,24-31}.
    v16h bmat[4];
#pragma unroll
    for (int dc = 0; dc < 4; ++dc) {
        v16h bm = {};
        if ((lane & 15) == 0) {
            int base = dc * 32 + ((lane < 16) ? 0 : 8);
#pragma unroll
            for (int i = 0; i < 16; ++i) bm[i] = qh[base + i + ((i < 8) ? 0 : 8)];
        }
        bmat[dc] = bm;
    }

    const int* selp = sel + ((size_t)b * H_ + h) * SEL_;
    const size_t page_stride = (size_t)(T_ * D_);

    // ---------------- Phase 1: QK scores (WMMA), TDM double-buffered -------
#if HAVE_TDM
    if (wave == 0) {
        int phys0 = btab[b * P_ + selp[0]];
        tdm_load_page(kc + ((size_t)phys0 * H_ + h) * page_stride, &kvbuf[0][0]);
    }
#endif
    for (int s = 0; s < SEL_; ++s) {
        const int* cur = kvbuf[s & 1];
#if HAVE_TDM
        if (wave == 0) {
            if (s + 1 < SEL_) {
                int physn = btab[b * P_ + selp[s + 1]];
                tdm_load_page(kc + ((size_t)physn * H_ + h) * page_stride,
                              &kvbuf[(s + 1) & 1][0]);
                __builtin_amdgcn_s_wait_tensorcnt(1);   // oldest (page s) done
            } else {
                __builtin_amdgcn_s_wait_tensorcnt(0);
            }
        }
        __syncthreads();
#else
        {
            int phys = btab[b * P_ + selp[s]];
            __syncthreads();
            vgpr_stage_page(kc + ((size_t)phys * H_ + h) * page_stride,
                            (int*)cur, tid);
            __syncthreads();
        }
#endif
        // A 16x32 f16 layout: lane<16 holds K{0-7,16-23}, lane>=16 K{8-15,24-31}
        int trow = wave * 16 + (lane & 15);
        v8f c = {};
#pragma unroll
        for (int dc = 0; dc < 4; ++dc) {
            int basei = trow * D_ + dc * 32 + ((lane < 16) ? 0 : 8);
            const int4* kp = (const int4*)&cur[basei];
            int4 i0 = kp[0];   // K +0..3
            int4 i1 = kp[1];   // K +4..7
            int4 i2 = kp[4];   // K +16..19
            int4 i3 = kp[5];   // K +20..23
            v16h a;
            a[0]  = (_Float16)(float)i0.x; a[1]  = (_Float16)(float)i0.y;
            a[2]  = (_Float16)(float)i0.z; a[3]  = (_Float16)(float)i0.w;
            a[4]  = (_Float16)(float)i1.x; a[5]  = (_Float16)(float)i1.y;
            a[6]  = (_Float16)(float)i1.z; a[7]  = (_Float16)(float)i1.w;
            a[8]  = (_Float16)(float)i2.x; a[9]  = (_Float16)(float)i2.y;
            a[10] = (_Float16)(float)i2.z; a[11] = (_Float16)(float)i2.w;
            a[12] = (_Float16)(float)i3.x; a[13] = (_Float16)(float)i3.y;
            a[14] = (_Float16)(float)i3.z; a[15] = (_Float16)(float)i3.w;
            c = __builtin_amdgcn_wmma_f32_16x16x32_f16(
                    false, a, false, bmat[dc], (short)0, c, false, false);
        }

        // column 0 of C: lane0 -> M=0..7 in c[0..7], lane16 -> M=8..15
        int p = selp[s];
        if ((lane & 15) == 0) {
            int mbase = wave * 16 + ((lane < 16) ? 0 : 8);
#pragma unroll
            for (int r = 0; r < 8; ++r) {
                int t   = mbase + r;
                int tok = p * T_ + t;
                att[s * T_ + t] = (tok < seqlen) ? c[r] * score_scale
                                                 : -__builtin_inff();
            }
        }
        __syncthreads();   // compute done before this buffer is refilled
    }

    // ---------------- softmax statistics over 2048 scores -------------------
    float m = -__builtin_inff();
    for (int i = tid; i < SEL_ * T_; i += 128) m = fmaxf(m, att[i]);
    red[tid] = m; __syncthreads();
    for (int s2 = 64; s2 > 0; s2 >>= 1) {
        if (tid < s2) red[tid] = fmaxf(red[tid], red[tid + s2]);
        __syncthreads();
    }
    m = red[0];
    __syncthreads();

    float psum = 0.0f;
    for (int i = tid; i < SEL_ * T_; i += 128) {
        float e = __expf(att[i] - m);
        att[i] = e;
        psum += e;
    }
    __syncthreads();
    red[tid] = psum; __syncthreads();
    for (int s2 = 64; s2 > 0; s2 >>= 1) {
        if (tid < s2) red[tid] += red[tid + s2];
        __syncthreads();
    }
    float denom = red[0];
    __syncthreads();

    // ---------------- Phase 2: weighted V accumulation, TDM streamed --------
    float acc = 0.0f;
    int d = tid;
#if HAVE_TDM
    if (wave == 0) {
        int phys0 = btab[b * P_ + selp[0]];
        tdm_load_page(vc + ((size_t)phys0 * H_ + h) * page_stride, &kvbuf[0][0]);
    }
#endif
    for (int s = 0; s < SEL_; ++s) {
        const int* cur = kvbuf[s & 1];
#if HAVE_TDM
        if (wave == 0) {
            if (s + 1 < SEL_) {
                int physn = btab[b * P_ + selp[s + 1]];
                tdm_load_page(vc + ((size_t)physn * H_ + h) * page_stride,
                              &kvbuf[(s + 1) & 1][0]);
                __builtin_amdgcn_s_wait_tensorcnt(1);
            } else {
                __builtin_amdgcn_s_wait_tensorcnt(0);
            }
        }
        __syncthreads();
#else
        {
            int phys = btab[b * P_ + selp[s]];
            __syncthreads();
            vgpr_stage_page(vc + ((size_t)phys * H_ + h) * page_stride,
                            (int*)cur, tid);
            __syncthreads();
        }
#endif
#pragma unroll 8
        for (int t = 0; t < T_; ++t)
            acc += att[s * T_ + t] * (float)cur[t * D_ + d];
        __syncthreads();
    }

    out[((size_t)b * H_ + h) * D_ + tid] = acc * vscale / denom;
}

// ---------------------------------------------------------------------------
extern "C" void kernel_launch(void* const* d_in, const int* in_sizes, int n_in,
                              void* d_out, int out_size, void* d_ws, size_t ws_size,
                              hipStream_t stream)
{
    const float* q    = (const float*)d_in[0];
    const int*   kc   = (const int*)d_in[1];
    const int*   vc   = (const int*)d_in[2];
    const int*   btab = (const int*)d_in[3];
    const int*   slen = (const int*)d_in[4];
    const float* ksc  = (const float*)d_in[5];
    const float* vsc  = (const float*)d_in[6];
    float*       out  = (float*)d_out;

    char* ws = (char*)d_ws;
    float* pm  = (float*)ws;                                           // 8 MB
    float* qrw = (float*)(ws + (size_t)NB_ * H_ * D_ * sizeof(float)); // 128 KB
    int*   sel = (int*)(ws + (size_t)NB_ * H_ * D_ * sizeof(float)
                            + (size_t)B_ * H_ * D_ * sizeof(float));

    hipLaunchKernelGGL(page_mean_kernel, dim3(NB_ * H_), dim3(D_), 0, stream, kc, pm);
    hipLaunchKernelGGL(select_kernel, dim3(B_ * H_), dim3(D_), 0, stream,
                       q, pm, btab, slen, qrw, sel);
    hipLaunchKernelGGL(attn_kernel, dim3(B_ * H_), dim3(D_), 0, stream,
                       kc, vc, btab, slen, qrw, sel, ksc, vsc, out);
}
